// MultiHeadMemoryBank_17961553232138
// MI455X (gfx1250) — matrix operand order repaired and checked by reading, not compiled, not run on hardware
//
#include <hip/hip_runtime.h>
#include <math.h>
#include <float.h>

#define BB  8
#define HH  8
#define NN  65536
#define DD  128
#define BDD 64
#define KK  32

typedef __attribute__((ext_vector_type(16))) _Float16 v16h;
typedef __attribute__((ext_vector_type(8)))  float    v8f;

__device__ __forceinline__ float sigmoidf_(float x){ return 1.0f/(1.0f+expf(-x)); }

// ---------------------------------------------------------------------------
// K1: normalize keys, sum of normalized read keys (for +1e-8 correction),
//     cvals = gelu(write_vals @ W_b1 + b_b1) @ W_b2 + b_b2   (exact gelu)
// grid: B*H blocks x 128 threads
// ---------------------------------------------------------------------------
__global__ __launch_bounds__(128) void k_prep(
    const float* __restrict__ wkeys, const float* __restrict__ rkeys,
    const float* __restrict__ wvals,
    const float* __restrict__ Wb1, const float* __restrict__ bb1,
    const float* __restrict__ Wb2, const float* __restrict__ bb2,
    float* __restrict__ wkn, float* __restrict__ rkn,
    float* __restrict__ Tr,  float* __restrict__ cvals)
{
  __shared__ float red[128];
  __shared__ float wv[128];
  __shared__ float hid[BDD];
  int bh = blockIdx.x, t = threadIdx.x;

  float wk = wkeys[bh*DD + t];
  red[t] = wk*wk; __syncthreads();
  for (int off=64; off>0; off>>=1){ if (t<off) red[t]+=red[t+off]; __syncthreads(); }
  float invw = 1.0f / fmaxf(sqrtf(red[0]), 1e-12f);
  __syncthreads();
  wkn[bh*DD + t] = wk*invw;

  float rk = rkeys[bh*DD + t];
  red[t] = rk*rk; __syncthreads();
  for (int off=64; off>0; off>>=1){ if (t<off) red[t]+=red[t+off]; __syncthreads(); }
  float invr = 1.0f / fmaxf(sqrtf(red[0]), 1e-12f);
  __syncthreads();
  float rkv = rk*invr;
  rkn[bh*DD + t] = rkv;

  red[t] = rkv; __syncthreads();
  for (int off=64; off>0; off>>=1){ if (t<off) red[t]+=red[t+off]; __syncthreads(); }
  if (t==0) Tr[bh] = red[0];
  __syncthreads();

  wv[t] = wvals[bh*DD + t]; __syncthreads();
  if (t < BDD){
    float a = bb1[t];
    for (int d=0; d<DD; ++d) a += wv[d]*Wb1[d*BDD + t];
    hid[t] = 0.5f*a*(1.0f + erff(a*0.70710678118654752f));   // exact gelu
  }
  __syncthreads();
  float c = bb2[t];
  for (int j=0; j<BDD; ++j) c += hid[j]*Wb2[j*DD + t];
  cvals[bh*DD + t] = c;
}

// ---------------------------------------------------------------------------
// K2: build A-matrix fragments (16 keys x 32 K-chunk) hi/lo f16 split in the
//     contiguous k order (k = 16*half + e) that B also uses. grid: B blocks x 32 threads.
//     Rows 0..7 = normalized write keys, 8..15 = normalized read keys.
// ---------------------------------------------------------------------------
__global__ __launch_bounds__(32) void k_afrag(
    const float* __restrict__ wkn, const float* __restrict__ rkn,
    _Float16* __restrict__ aHi, _Float16* __restrict__ aLo)
{
  int b = blockIdx.x, lane = threadIdx.x;
  int m = lane & 15, hs = lane >> 4;
  const float* key = (m < HH) ? (wkn + (b*HH + m)*DD) : (rkn + (b*HH + (m-HH))*DD);
  for (int c=0; c<4; ++c){
    for (int e=0; e<16; ++e){
      // A 16x32 f16 layout (contiguous, matching B): lanes 0-15: K=e
      //                                              lanes16-31: K=16+e
      int kk = c*32 + hs*16 + e;
      float x = key[kk];
      _Float16 hi = (_Float16)x;
      _Float16 lo = (_Float16)(x - (float)hi);
      size_t o = ((size_t)(b*4 + c)*32 + lane)*16 + (size_t)e;
      aHi[o] = hi; aLo[o] = lo;
    }
  }
}

// ---------------------------------------------------------------------------
// K3: single streaming pass over memory. Each wave handles 16 rows:
//     12x v_wmma_f32_16x16x32_f16 (hi/lo compensated) -> 16 keys x 16 rows.
//     Also per-row norm^2 and sum (shfl_xor 16 to combine half-rows).
//     Writes sim_w (beta_w applied) and sim_r approx (+1e-8 corrected).
// ---------------------------------------------------------------------------
__global__ __launch_bounds__(256) void k_sim(
    const float* __restrict__ mem,
    const _Float16* __restrict__ aHi, const _Float16* __restrict__ aLo,
    const float* __restrict__ Tr,
    const float* __restrict__ beta_w, const float* __restrict__ beta_r,
    float* __restrict__ simw, float* __restrict__ simr)
{
  const int TPB_ = NN/16;
  const int TOT  = BB*TPB_;
  int gid  = blockIdx.x*blockDim.x + threadIdx.x;
  int wv_  = gid >> 5;
  int nw   = (gridDim.x*blockDim.x) >> 5;
  int lane = threadIdx.x & 31;
  int col  = lane & 15, hs = lane >> 4;

  for (int tile = wv_; tile < TOT; tile += nw){
    int b  = tile / TPB_;
    int tn = tile % TPB_;
    int nIdx = tn*16 + col;
    // B 32x16 f16 layout: lane L holds column L%16, K = (L/16)*16 + e : contiguous!
    const float* rowp = mem + ((size_t)b*NN + nIdx)*DD + hs*16;

    int nt = tile + nw;                       // prefetch next tile (global_prefetch_b8)
    if (nt < TOT){
      int b2 = nt / TPB_, tn2 = nt % TPB_;
      __builtin_prefetch(mem + ((size_t)b2*NN + tn2*16 + col)*DD + hs*16, 0, 1);
    }

    float nrm2 = 0.f, rsum = 0.f;
    v8f acc = {0.f,0.f,0.f,0.f,0.f,0.f,0.f,0.f};
    #pragma unroll
    for (int c=0; c<4; ++c){
      const float4* f4 = reinterpret_cast<const float4*>(rowp + c*32);
      v16h bhi, blo;
      #pragma unroll
      for (int q=0; q<4; ++q){
        float4 f = f4[q];
        float xs[4] = {f.x, f.y, f.z, f.w};
        #pragma unroll
        for (int e=0; e<4; ++e){
          float x = xs[e];
          nrm2 += x*x; rsum += x;
          _Float16 h = (_Float16)x;
          bhi[q*4+e] = h;
          blo[q*4+e] = (_Float16)(x - (float)h);
        }
      }
      size_t ao = ((size_t)(b*4 + c)*32 + lane)*16;
      v16h ah = *reinterpret_cast<const v16h*>(aHi + ao);
      v16h al = *reinterpret_cast<const v16h*>(aLo + ao);
      acc = __builtin_amdgcn_wmma_f32_16x16x32_f16(false, al, false, bhi, (short)0, acc, false, false);
      acc = __builtin_amdgcn_wmma_f32_16x16x32_f16(false, ah, false, blo, (short)0, acc, false, false);
      acc = __builtin_amdgcn_wmma_f32_16x16x32_f16(false, ah, false, bhi, (short)0, acc, false, false);
    }
    nrm2 += __shfl_xor(nrm2, 16);
    rsum += __shfl_xor(rsum, 16);
    float inv = 1.0f / fmaxf(sqrtf(nrm2), 1e-12f);
    if (hs == 0){                                  // C rows 0..7 = write heads
      #pragma unroll
      for (int v=0; v<8; ++v)
        simw[(size_t)(b*HH+v)*NN + nIdx] = acc[v]*inv*beta_w[b*HH+v];
    } else {                                       // C rows 8..15 = read heads
      float n2e  = nrm2 + 2e-8f*rsum + (float)DD*1e-16f;   // ||mem+1e-8||^2
      float inve = 1.0f / fmaxf(sqrtf(n2e), 1e-12f);
      #pragma unroll
      for (int v=0; v<8; ++v){
        float d = acc[v] + 1e-8f*Tr[b*HH+v];
        simr[(size_t)(b*HH+v)*NN + nIdx] = d*inve*beta_r[b*HH+v];
      }
    }
  }
}

// ---------------------------------------------------------------------------
// K4: top-32 + softmax per (b,h): 2048-bin radix histogram over monotonic
//     float keys, candidate collect, 32x tie-broken argmax. grid: 64 x 256.
// ---------------------------------------------------------------------------
#define NBIN 2048
#define CAP  1024
__device__ __forceinline__ unsigned okey(float x){
  unsigned u = __float_as_uint(x);
  return (u & 0x80000000u) ? ~u : (u | 0x80000000u);
}
__global__ __launch_bounds__(256) void k_topk(
    const float* __restrict__ sim, int* __restrict__ oIdx, float* __restrict__ oW)
{
  __shared__ unsigned hist[NBIN];
  __shared__ float cv[CAP]; __shared__ int ci[CAP];
  __shared__ float rv[256]; __shared__ int ri[256]; __shared__ int rs[256];
  __shared__ float selV[KK]; __shared__ int selI[KK];
  __shared__ int binSel; __shared__ int cnt;
  int bh = blockIdx.x, t = threadIdx.x;
  const float* s = sim + (size_t)bh*NN;

  for (int i=t; i<NBIN; i+=256) hist[i]=0;
  if (t==0) cnt=0;
  __syncthreads();
  for (int i=t; i<NN; i+=256) atomicAdd(&hist[okey(s[i])>>21], 1u);
  __syncthreads();
  if (t==0){
    unsigned cum=0; int bsel=0;
    for (int i=NBIN-1; i>=0; --i){ cum+=hist[i]; if (cum>=KK){ bsel=i; break; } }
    binSel=bsel;
  }
  __syncthreads();
  unsigned bs = (unsigned)binSel;
  for (int i=t; i<NN; i+=256){
    if ((okey(s[i])>>21) >= bs){
      int p = atomicAdd(&cnt,1);
      if (p < CAP){ cv[p]=s[i]; ci[p]=i; }
    }
  }
  __syncthreads();
  int M = cnt; if (M > CAP) M = CAP;
  for (int j=0; j<KK; ++j){
    float bv=-FLT_MAX; int bi=0x7fffffff, bslot=-1;
    for (int i=t; i<M; i+=256){
      float v=cv[i]; int id=ci[i];
      if (v>bv || (v==bv && id<bi)){ bv=v; bi=id; bslot=i; }
    }
    rv[t]=bv; ri[t]=bi; rs[t]=bslot;
    __syncthreads();
    for (int off=128; off>0; off>>=1){
      if (t<off){
        if (rv[t+off]>rv[t] || (rv[t+off]==rv[t] && ri[t+off]<ri[t])){
          rv[t]=rv[t+off]; ri[t]=ri[t+off]; rs[t]=rs[t+off];
        }
      }
      __syncthreads();
    }
    if (t==0){ selV[j]=rv[0]; selI[j]=ri[0]; if (rs[0]>=0) cv[rs[0]]=-FLT_MAX; }
    __syncthreads();
  }
  if (t < KK){
    float m = selV[0];
    for (int j=1; j<KK; ++j) m = fmaxf(m, selV[j]);
    float sum=0.f; for (int j=0; j<KK; ++j) sum += expf(selV[j]-m);
    oW[bh*KK + t]   = expf(selV[t]-m)/sum;
    oIdx[bh*KK + t] = selI[t];
  }
}

// ---------------------------------------------------------------------------
// K5: sparse erase/add update on the <=256 touched rows per batch; stores
//     mem2 rows (decayed) compactly and overwrites sim_r exactly for them.
// grid: B x 256
// ---------------------------------------------------------------------------
__global__ __launch_bounds__(256) void k_update(
    const float* __restrict__ mem,
    const int* __restrict__ widx, const float* __restrict__ wW,
    const float* __restrict__ erase, const float* __restrict__ addg,
    const float* __restrict__ cvals, const float* __restrict__ rkn,
    const float* __restrict__ beta_r, const float* __restrict__ decay,
    float* __restrict__ simr,
    int* __restrict__ updCount, int* __restrict__ updN, float* __restrict__ updRows)
{
  __shared__ int   en[256]; __shared__ float ewc[256], eac[256];
  __shared__ int   un[256]; __shared__ float uem[256]; __shared__ float uac[256*8];
  __shared__ int   cnt;
  __shared__ float red[256]; __shared__ float urow[128];
  int b = blockIdx.x, t = threadIdx.x;
  int h = t >> 5;
  int n0 = widx[(b*HH+h)*KK + (t&31)];
  float w = wW [(b*HH+h)*KK + (t&31)];
  en[t]=n0; ewc[t]=w*erase[b*HH+h]*(1.0f/HH); eac[t]=w*addg[b*HH+h]*(1.0f/HH);
  if (t==0) cnt=0;
  __syncthreads();
  bool uniq=true;
  for (int j=0; j<t; ++j) if (en[j]==n0){ uniq=false; break; }
  if (uniq){
    int slot = atomicAdd(&cnt,1);
    un[slot] = n0;
    float em=0.f, ac[8]={0,0,0,0,0,0,0,0};
    for (int j=0; j<256; ++j) if (en[j]==n0){ em+=ewc[j]; ac[j>>5]+=eac[j]; }
    uem[slot]=em;
    for (int h2=0; h2<8; ++h2) uac[slot*8+h2]=ac[h2];
  }
  __syncthreads();
  int C = cnt;
  for (int ss=0; ss<C; ++ss){
    int n = un[ss];
    float v = 0.f;
    if (t < DD){
      v = mem[((size_t)b*NN + n)*DD + t]*(1.0f - uem[ss]);
      for (int h2=0; h2<8; ++h2) v += uac[ss*8+h2]*cvals[(b*HH+h2)*DD + t];
      v += 1e-8f;
    }
    red[t] = (t<DD)? v*v : 0.f;
    __syncthreads();
    for (int off=128; off>0; off>>=1){ if (t<off) red[t]+=red[t+off]; __syncthreads(); }
    float inv = 1.0f/fmaxf(sqrtf(red[0]), 1e-12f);
    __syncthreads();
    if (t < DD){
      float u = v*inv;
      urow[t] = u;
      updRows[((size_t)b*256 + ss)*DD + t] = u * sigmoidf_(decay[n]);
    }
    __syncthreads();
    {   // exact sim_r overwrite: wave h computes head h dot via shfl reduce
      int wh = t>>5, lane = t&31;
      float p = 0.f;
      for (int q=0; q<4; ++q){ int d = lane + 32*q; p += rkn[(b*HH+wh)*DD + d]*urow[d]; }
      for (int off=16; off>0; off>>=1) p += __shfl_xor(p, off);
      if (lane==0) simr[(size_t)(b*HH+wh)*NN + n] = p*beta_r[b*HH+wh];
    }
    __syncthreads();
  }
  if (t==0) updCount[b]=C;
  if (t < C) updN[b*256 + t] = un[t];
}

// ---------------------------------------------------------------------------
// K6: gather rph from mem2 (stored rows or recomputed untouched rows),
//     merge GEMV (1024x128) + LayerNorm -> out[b,128]. grid: B x 256
// ---------------------------------------------------------------------------
__global__ __launch_bounds__(256) void k_readout(
    const float* __restrict__ mem,
    const int* __restrict__ ridx, const float* __restrict__ rW,
    const int* __restrict__ updCount, const int* __restrict__ updN,
    const float* __restrict__ updRows, const float* __restrict__ decay,
    const float* __restrict__ Wm, const float* __restrict__ bm,
    const float* __restrict__ lng, const float* __restrict__ lnb,
    float* __restrict__ out)
{
  __shared__ float rph[HH*DD];
  __shared__ int   uN[256];
  __shared__ float red[256];
  __shared__ float mg[128];
  __shared__ int sn; __shared__ int sslot; __shared__ float sw;
  int b = blockIdx.x, t = threadIdx.x;
  int C = updCount[b];
  for (int i=t; i<HH*DD; i+=256) rph[i]=0.f;
  if (t < C) uN[t] = updN[b*256 + t];
  __syncthreads();
  for (int p=0; p<HH*KK; ++p){
    if (t==0){
      int n = ridx[b*HH*KK + p];
      int sl=-1;
      for (int s2=0; s2<C; ++s2) if (uN[s2]==n){ sl=s2; break; }
      sn=n; sslot=sl; sw=rW[b*HH*KK + p];
    }
    __syncthreads();
    int n=sn, sl=sslot; float w=sw;
    float r = 0.f;
    if (sl >= 0){                                // updated row: stored exactly
      if (t<DD) r = updRows[((size_t)b*256 + sl)*DD + t];
    } else {                                     // untouched: unit(mem+1e-8)*sig(decay)
      float v = (t<DD)? (mem[((size_t)b*NN + n)*DD + t] + 1e-8f) : 0.f;
      red[t] = v*v;
      __syncthreads();
      for (int off=128; off>0; off>>=1){ if (t<off) red[t]+=red[t+off]; __syncthreads(); }
      float inv = 1.0f/fmaxf(sqrtf(red[0]), 1e-12f);
      __syncthreads();
      if (t<DD) r = v*inv*sigmoidf_(decay[n]);
    }
    if (t<DD) rph[(p>>5)*DD + t] += w*r;         // head = p/32
    __syncthreads();
  }
  float m = 0.f;
  if (t < DD){
    m = bm[t];
    for (int j=0; j<HH*DD; ++j) m += rph[j]*Wm[j*DD + t];
    mg[t] = m;
  }
  red[t] = (t<DD)? m : 0.f;
  __syncthreads();
  for (int off=128; off>0; off>>=1){ if (t<off) red[t]+=red[t+off]; __syncthreads(); }
  float mu = red[0]*(1.0f/DD);
  __syncthreads();
  float dv = (t<DD)? (mg[t]-mu) : 0.f;
  red[t] = dv*dv;
  __syncthreads();
  for (int off=128; off>0; off>>=1){ if (t<off) red[t]+=red[t+off]; __syncthreads(); }
  float var = red[0]*(1.0f/DD);
  if (t < DD) out[b*DD + t] = (mg[t]-mu)*rsqrtf(var+1e-5f)*lng[t] + lnb[t];
}

// ---------------------------------------------------------------------------
extern "C" void kernel_launch(void* const* d_in, const int* in_sizes, int n_in,
                              void* d_out, int out_size, void* d_ws, size_t ws_size,
                              hipStream_t stream)
{
  (void)in_sizes; (void)n_in; (void)out_size; (void)ws_size;
  const float* memory     = (const float*)d_in[0];
  const float* read_keys  = (const float*)d_in[1];
  const float* write_keys = (const float*)d_in[2];
  const float* write_vals = (const float*)d_in[3];
  const float* erase      = (const float*)d_in[4];
  const float* add_gate   = (const float*)d_in[5];
  const float* beta_r     = (const float*)d_in[6];
  const float* beta_w     = (const float*)d_in[7];
  const float* W_b1       = (const float*)d_in[8];
  const float* b_b1       = (const float*)d_in[9];
  const float* W_b2       = (const float*)d_in[10];
  const float* b_b2       = (const float*)d_in[11];
  const float* W_merge    = (const float*)d_in[12];
  const float* b_merge    = (const float*)d_in[13];
  const float* ln_g       = (const float*)d_in[14];
  const float* ln_b       = (const float*)d_in[15];
  const float* decay      = (const float*)d_in[16];
  float* out = (float*)d_out;

  char* ws = (char*)d_ws;
  size_t off = 0;
  auto alloc = [&](size_t bytes)->char* {
    char* p = ws + off; off = (off + bytes + 255) & ~(size_t)255; return p;
  };
  float*    wkn      = (float*)   alloc((size_t)BB*HH*DD*4);
  float*    rkn      = (float*)   alloc((size_t)BB*HH*DD*4);
  float*    Tr       = (float*)   alloc((size_t)BB*HH*4);
  float*    cvals    = (float*)   alloc((size_t)BB*HH*DD*4);
  _Float16* aHi      = (_Float16*)alloc((size_t)BB*4*32*16*2);
  _Float16* aLo      = (_Float16*)alloc((size_t)BB*4*32*16*2);
  float*    simw     = (float*)   alloc((size_t)BB*HH*NN*4);
  float*    simr     = (float*)   alloc((size_t)BB*HH*NN*4);
  int*      widx     = (int*)     alloc((size_t)BB*HH*KK*4);
  float*    wW       = (float*)   alloc((size_t)BB*HH*KK*4);
  int*      ridx     = (int*)     alloc((size_t)BB*HH*KK*4);
  float*    rW       = (float*)   alloc((size_t)BB*HH*KK*4);
  int*      updCnt   = (int*)     alloc((size_t)BB*4);
  int*      updN     = (int*)     alloc((size_t)BB*256*4);
  float*    updRows  = (float*)   alloc((size_t)BB*256*DD*4);

  k_prep   <<<BB*HH, 128, 0, stream>>>(write_keys, read_keys, write_vals,
                                       W_b1, b_b1, W_b2, b_b2, wkn, rkn, Tr, cvals);
  k_afrag  <<<BB,     32, 0, stream>>>(wkn, rkn, aHi, aLo);
  k_sim    <<<512,   256, 0, stream>>>(memory, aHi, aLo, Tr, beta_w, beta_r, simw, simr);
  k_topk   <<<BB*HH, 256, 0, stream>>>(simw, widx, wW);
  k_update <<<BB,    256, 0, stream>>>(memory, widx, wW, erase, add_gate, cvals, rkn,
                                       beta_r, decay, simr, updCnt, updN, updRows);
  k_topk   <<<BB*HH, 256, 0, stream>>>(simr, ridx, rW);
  k_readout<<<BB,    256, 0, stream>>>(memory, ridx, rW, updCnt, updN, updRows, decay,
                                       W_merge, b_merge, ln_g, ln_b, out);
}
